// FedFPNNR_70342974374102
// MI455X (gfx1250) — matrix operand order repaired and checked by reading, not compiled, not run on hardware
//
#include <hip/hip_runtime.h>
#include <math.h>

// ---------------------------------------------------------------------------
// FedFPNN forward for MI455X (gfx1250, wave32, WMMA).
// Compute-bound (~73 GFLOP vs ~11 MB effective traffic) -> bf16 WMMA path.
// ---------------------------------------------------------------------------

typedef __bf16 bf16_t;
typedef __attribute__((ext_vector_type(8)))  __bf16 v8bf;
typedef __attribute__((ext_vector_type(16))) __bf16 v16bf;
typedef __attribute__((ext_vector_type(8)))  float  v8f;

#define N_FEA    256
#define N_CLASS  64
#define N_RULE   32
#define BATCH    8192
#define N_ACTIVE 16
#define F2       512   // 2*N_FEA

__device__ __forceinline__ float elu_f(float x) {
    return x > 0.0f ? x : (__expf(x) - 1.0f);
}

// A-fragment (16x32 bf16, M x K) from an LDS row-major tile, contiguous k order (same as load_b_frag):
// lanes 0-15 hold row (lane), K = k0+0..15;
// lanes 16-31 hold row (lane-16), K = k0+16..31.
__device__ __forceinline__ v16bf load_a_frag(const bf16_t* __restrict__ base,
                                             int ldk, int m0, int k0, int lane) {
    const int half = lane >> 4;
    const int row  = m0 + (lane & 15);
    const bf16_t* p = base + row * ldk + k0 + half * 16;
    v8bf lo = *(const v8bf*)(p);          // 16B LDS/global load
    v8bf hi = *(const v8bf*)(p + 8);      // 16B
    return __builtin_shufflevector(lo, hi, 0,1,2,3,4,5,6,7,8,9,10,11,12,13,14,15);
}

// B-fragment (32x16 bf16, K x N) where B[k][n] = W[n][k] (row-major weight,
// row length ldk). Column n of B == contiguous weight row -> one 32B load.
// lanes 0-15: n=lane, K=k0..k0+15 ; lanes 16-31: n=lane-16, K=k0+16..k0+31.
__device__ __forceinline__ v16bf load_b_frag(const bf16_t* __restrict__ wrow,
                                             int ldk, int n0, int k0, int lane) {
    const int half = lane >> 4;
    const int col  = n0 + (lane & 15);
    return *(const v16bf*)(wrow + col * ldk + k0 + half * 16);
}

// ---------------------------------------------------------------------------
// Kernel 0: fp32 -> bf16 weight conversion into workspace
// ---------------------------------------------------------------------------
__global__ void cvt_kernel(const float* __restrict__ src, bf16_t* __restrict__ dst, int n) {
    int i = blockIdx.x * blockDim.x + threadIdx.x;
    if (i < n) dst[i] = (bf16_t)src[i];
}

// ---------------------------------------------------------------------------
// Kernel A: per (batch-tile of 32, active rule k): membership -> MLP -> fsi
//   mv  = exp(-(x-proto)^2 / (2 v^2))            [32,256]  (VALU, bf16 to LDS)
//   h1  = ELU(mv @ W1^T + b1)                     [32,512]  (WMMA, bf16 to LDS)
//   h2  = ELU(h1 @ W2^T + b2); fsi = h2.W3 + b3   [32]      (WMMA + ds_add_f32)
// ---------------------------------------------------------------------------
__global__ void __launch_bounds__(256) fsi_kernel(
    const float* __restrict__ x,     const float* __restrict__ proto,
    const float* __restrict__ var,
    const bf16_t* __restrict__ W1b,  const float* __restrict__ b1,
    const bf16_t* __restrict__ W2b,  const float* __restrict__ b2,
    const float* __restrict__ W3,    const float* __restrict__ b3,
    const int* __restrict__ rules_idx, float* __restrict__ fsi_act)
{
    __shared__ bf16_t As[32 * N_FEA];   // 16 KB  membership tile
    __shared__ bf16_t Hs[32 * F2];      // 32 KB  h1 tile
    __shared__ float  sfsi[32];

    const int tid  = threadIdx.x;
    const int lane = tid & 31;
    const int w    = tid >> 5;          // 8 waves
    const int b0   = blockIdx.x * 32;
    const int kidx = blockIdx.y;
    const int r    = rules_idx[kidx];

    if (tid < 32) sfsi[tid] = 0.0f;

    // ---- membership (fp32 exp, bf16 into LDS) ----
    for (int idx = tid; idx < 32 * N_FEA; idx += 256) {
        const int bl = idx >> 8, f = idx & 255;
        float xv = x[(b0 + bl) * N_FEA + f];
        float pv = proto[r * N_FEA + f];
        float vv = var[r * N_FEA + f];
        vv = fminf(fmaxf(vv, 1e-4f), 0.1f);
        float d  = xv - pv;
        As[idx]  = (bf16_t)__expf(-(d * d) / (2.0f * vv * vv));
    }
    __syncthreads();

    // ---- GEMM1: 32 x 512 x 256, 64 tiles over 8 waves ----
    for (int t = w; t < 64; t += 8) {
        const int mt = (t & 1) * 16;
        const int nt = (t >> 1) * 16;
        v8f acc = {};
        #pragma unroll
        for (int kt = 0; kt < N_FEA; kt += 32) {
            v16bf a = load_a_frag(As, N_FEA, mt, kt, lane);
            v16bf b = load_b_frag(W1b, N_FEA, nt, kt, lane);
            acc = __builtin_amdgcn_wmma_f32_16x16x32_bf16(
                false, a, false, b, (short)0, acc, false, false);
        }
        const int col     = nt + (lane & 15);
        const int rowbase = mt + ((lane >> 4) << 3);
        const float bias  = b1[col];
        #pragma unroll
        for (int i = 0; i < 8; ++i)
            Hs[(rowbase + i) * F2 + col] = (bf16_t)elu_f(acc[i] + bias);
    }
    __syncthreads();

    // ---- GEMM2 + fused W3 dot: 32 x 256 x 512, 32 tiles over 8 waves ----
    for (int t = w; t < 32; t += 8) {
        const int mt = (t & 1) * 16;
        const int nt = (t >> 1) * 16;
        v8f acc = {};
        #pragma unroll 4
        for (int kt = 0; kt < F2; kt += 32) {
            v16bf a = load_a_frag(Hs, F2, mt, kt, lane);
            v16bf b = load_b_frag(W2b, F2, nt, kt, lane);
            acc = __builtin_amdgcn_wmma_f32_16x16x32_bf16(
                false, a, false, b, (short)0, acc, false, false);
        }
        const int col     = nt + (lane & 15);
        const int rowbase = mt + ((lane >> 4) << 3);
        const float bias  = b2[col];
        const float w3v   = W3[col];
        #pragma unroll
        for (int i = 0; i < 8; ++i)
            atomicAdd(&sfsi[rowbase + i], elu_f(acc[i] + bias) * w3v);
    }
    __syncthreads();

    if (tid < 32)
        fsi_act[(b0 + tid) * N_ACTIVE + kidx] = sfsi[tid] + b3[0];
}

// ---------------------------------------------------------------------------
// Kernel B: softmax over the K=16 active-rule columns, one row per thread
// ---------------------------------------------------------------------------
__global__ void softmax_kernel(const float* __restrict__ fsi_act,
                               float* __restrict__ fire)
{
    const int b = blockIdx.x * blockDim.x + threadIdx.x;
    if (b >= BATCH) return;
    float v[N_ACTIVE];
    float mx = -INFINITY;
    #pragma unroll
    for (int k = 0; k < N_ACTIVE; ++k) {
        v[k] = fsi_act[b * N_ACTIVE + k];
        mx = fmaxf(mx, v[k]);
    }
    float s = 0.0f;
    #pragma unroll
    for (int k = 0; k < N_ACTIVE; ++k) { v[k] = __expf(v[k] - mx); s += v[k]; }
    const float inv = 1.0f / s;
    #pragma unroll
    for (int k = 0; k < N_ACTIVE; ++k) fire[b * N_ACTIVE + k] = v[k] * inv;
}

// ---------------------------------------------------------------------------
// Kernel C: outputs[b][c] = sum_k fire[b][k] * relu(x[b].Wc[rk][c] + bc[rk][c])
// 32x64 output tile per block; 8 waves each own one 16x16 tile across all k.
// ---------------------------------------------------------------------------
__global__ void __launch_bounds__(256) consequent_kernel(
    const float* __restrict__ x, const bf16_t* __restrict__ Wcb,
    const float* __restrict__ bc, const int* __restrict__ rules_idx,
    const float* __restrict__ fire, float* __restrict__ out)
{
    __shared__ bf16_t Xs[32 * N_FEA];     // 16 KB
    __shared__ float  Fs[32 * N_ACTIVE];  // 2 KB

    const int tid  = threadIdx.x;
    const int lane = tid & 31;
    const int w    = tid >> 5;
    const int b0   = blockIdx.x * 32;

    for (int idx = tid; idx < 32 * N_FEA; idx += 256)
        Xs[idx] = (bf16_t)x[(b0 + (idx >> 8)) * N_FEA + (idx & 255)];
    for (int idx = tid; idx < 32 * N_ACTIVE; idx += 256)
        Fs[idx] = fire[b0 * N_ACTIVE + idx];
    __syncthreads();

    const int mt      = (w >> 2) * 16;          // 0 / 16
    const int nt      = (w & 3) * 16;           // 0..48
    const int col     = nt + (lane & 15);
    const int rowbase = mt + ((lane >> 4) << 3);

    v8f oacc = {};
    for (int k = 0; k < N_ACTIVE; ++k) {
        const int r = rules_idx[k];
        const bf16_t* Wr = Wcb + r * (N_CLASS * N_FEA);
        v8f acc = {};
        #pragma unroll
        for (int kt = 0; kt < N_FEA; kt += 32) {
            v16bf a = load_a_frag(Xs, N_FEA, mt, kt, lane);
            v16bf b = load_b_frag(Wr, N_FEA, nt, kt, lane);
            acc = __builtin_amdgcn_wmma_f32_16x16x32_bf16(
                false, a, false, b, (short)0, acc, false, false);
        }
        const float bias = bc[r * N_CLASS + col];
        #pragma unroll
        for (int i = 0; i < 8; ++i) {
            float dpv = fmaxf(acc[i] + bias, 0.0f);
            oacc[i] += dpv * Fs[(rowbase + i) * N_ACTIVE + k];
        }
    }
    #pragma unroll
    for (int i = 0; i < 8; ++i)
        out[(b0 + rowbase + i) * N_CLASS + col] = oacc[i];
}

// ---------------------------------------------------------------------------
// Launch
// ---------------------------------------------------------------------------
extern "C" void kernel_launch(void* const* d_in, const int* in_sizes, int n_in,
                              void* d_out, int out_size, void* d_ws, size_t ws_size,
                              hipStream_t stream) {
    const float* x     = (const float*)d_in[0];
    const float* proto = (const float*)d_in[1];
    const float* var   = (const float*)d_in[2];
    const float* W1    = (const float*)d_in[3];
    const float* b1    = (const float*)d_in[4];
    const float* W2    = (const float*)d_in[5];
    const float* b2    = (const float*)d_in[6];
    const float* W3    = (const float*)d_in[7];
    const float* b3    = (const float*)d_in[8];
    const float* Wc    = (const float*)d_in[9];
    const float* bc    = (const float*)d_in[10];
    const int*   ridx  = (const int*)d_in[11];

    // workspace layout (bytes)
    char* ws = (char*)d_ws;
    bf16_t* W1b    = (bf16_t*)(ws);                       // 512*256*2 = 256 KB
    bf16_t* W2b    = (bf16_t*)(ws + (256u << 10));        // 256*512*2 = 256 KB
    bf16_t* Wcb    = (bf16_t*)(ws + (512u << 10));        // 32*64*256*2 = 1 MB
    float*  fsiact = (float*)(ws + (512u << 10) + (1024u << 10)); // 8192*16*4

    float* out  = (float*)d_out;                 // [BATCH, N_CLASS]
    float* fire = out + (size_t)BATCH * N_CLASS; // [BATCH, N_ACTIVE]

    cvt_kernel<<<(F2 * N_FEA + 255) / 256, 256, 0, stream>>>(W1, W1b, F2 * N_FEA);
    cvt_kernel<<<(N_FEA * F2 + 255) / 256, 256, 0, stream>>>(W2, W2b, N_FEA * F2);
    cvt_kernel<<<(N_RULE * N_CLASS * N_FEA + 255) / 256, 256, 0, stream>>>(
        Wc, Wcb, N_RULE * N_CLASS * N_FEA);

    fsi_kernel<<<dim3(BATCH / 32, N_ACTIVE), 256, 0, stream>>>(
        x, proto, var, W1b, b1, W2b, b2, W3, b3, ridx, fsiact);

    softmax_kernel<<<BATCH / 256, 256, 0, stream>>>(fsiact, fire);

    consequent_kernel<<<BATCH / 32, 256, 0, stream>>>(x, Wcb, bc, ridx, fire, out);
}
